// Model_87222195847183
// MI455X (gfx1250) — compile-verified, matrix operand order checked by reading, not run on hardware
//
#include <hip/hip_runtime.h>
#include <hip/hip_bf16.h>
#include <math.h>

// ---------------------------------------------------------------------------
// Types for CDNA5 WMMA (wave32)
// ---------------------------------------------------------------------------
typedef __attribute__((ext_vector_type(16))) __bf16        v16bf;
typedef __attribute__((ext_vector_type(8)))  float         v8f;
typedef __attribute__((ext_vector_type(8)))  unsigned short ush8;

union BFrag { v16bf f; ush8 h[2]; };

__device__ __forceinline__ float act_apply(float v, int act) {
    switch (act) {
        case 1: return v > 0.f ? v : 0.f;                    // relu
        case 2: return v > 0.f ? v : 0.1f * v;               // leaky relu 0.1
        case 3: return tanhf(v);                             // tanh
        case 4: return v > 20.f ? v : log1pf(expf(v));       // softplus
        default: return v;
    }
}

// CDNA5 async global->LDS DMA (ASYNCcnt); lds_off = byte offset in LDS space
__device__ __forceinline__ void async_b128(const unsigned short* g, unsigned lds_off) {
    asm volatile("global_load_async_to_lds_b128 %0, %1, off"
                 :: "v"(lds_off), "v"(g) : "memory");
}

// ---------------------------------------------------------------------------
// Tiled WMMA GEMM: C(MxN,f32) = act( A(MxK,bf16) @ Bt(NxK,bf16)^T + Cadd + bias )
// Block = 256 threads (8 waves) computing a 128x128 C tile.
// Waves arranged 4(M) x 2(N); each wave owns 32x64 = 2x4 16x16 accumulators.
// A/B K-tiles (128x32 bf16, 8KB each) staged in LDS with double-buffered
// GLOBAL_LOAD_ASYNC_TO_LDS_B128; fragments then read via ds_load_b128.
// Fragment layout per CDNA5 ISA 7.12.2 (16-bit A 16x32): lane<16 holds row
// M=lane, K={k0..k0+7, k0+16..k0+23}; lane>=16 K shifted by 8. B from Bt rows.
// Requires M%128==0, N%128==0, K%32==0.
// ---------------------------------------------------------------------------
#define BM 128
#define BN 128
#define BK 32

__global__ void __launch_bounds__(256)
wmma_gemm_tiled_kernel(const unsigned short* __restrict__ A,
                       const unsigned short* __restrict__ Bt,
                       const float* __restrict__ bias,
                       const float* __restrict__ Cadd,
                       float* __restrict__ C,
                       int M, int N, int K, int act)
{
    __shared__ __align__(16) unsigned short shA[2][BM * BK];
    __shared__ __align__(16) unsigned short shB[2][BN * BK];

    int tilesN = N / BN;
    int bm = blockIdx.x / tilesN;
    int bn = blockIdx.x - bm * tilesN;
    int tid  = threadIdx.x;
    int w    = tid >> 5;
    int lane = tid & 31;
    int half = lane >> 4;
    int lr   = lane & 15;
    int wm   = w >> 1;          // 0..3
    int wn   = w & 1;           // 0..1

    const unsigned short* Abase = A  + (size_t)(bm * BM) * K;
    const unsigned short* Bbase = Bt + (size_t)(bn * BN) * K;

    // LDS byte addresses of the staging buffers (low 32 bits of generic ptr)
    unsigned ldsA = (unsigned)(size_t)&shA[0][0];
    unsigned ldsB = (unsigned)(size_t)&shB[0][0];

    int NKc = K / BK;

    auto stage_load = [&](int st, int kc) {
        // 8KB tile = 512 x 16B transfers; 256 threads x 2 each; 4 async
        // instructions per wave per stage (2 for A + 2 for B).
#pragma unroll
        for (int t = 0; t < 2; ++t) {
            int j = tid + t * 256;
            int row = j >> 2, seg = j & 3;
            async_b128(Abase + (size_t)row * K + kc * BK + seg * 8,
                       ldsA + (unsigned)(st * BM * BK + row * BK + seg * 8) * 2u);
        }
#pragma unroll
        for (int t = 0; t < 2; ++t) {
            int j = tid + t * 256;
            int row = j >> 2, seg = j & 3;
            async_b128(Bbase + (size_t)row * K + kc * BK + seg * 8,
                       ldsB + (unsigned)(st * BN * BK + row * BK + seg * 8) * 2u);
        }
    };

    v8f acc[2][4] = {};
    stage_load(0, 0);

    for (int kc = 0; kc < NKc; ++kc) {
        int cur = kc & 1;
        if (kc + 1 < NKc) {
            stage_load(1 - cur, kc + 1);
            // 4 async loads of stage kc+1 outstanding; <=4 => stage kc resident
            asm volatile("s_wait_asynccnt 0x4" ::: "memory");
        } else {
            asm volatile("s_wait_asynccnt 0x0" ::: "memory");
        }
        __syncthreads();          // publish all waves' DMA fills

        BFrag af[2], bf4[4];
#pragma unroll
        for (int mi = 0; mi < 2; ++mi) {
            const unsigned short* p = &shA[cur][(wm * 32 + mi * 16 + lr) * BK];
            af[mi].h[0] = *(const ush8*)(p + half * 8);
            af[mi].h[1] = *(const ush8*)(p + 16 + half * 8);
        }
#pragma unroll
        for (int ni = 0; ni < 4; ++ni) {
            const unsigned short* p = &shB[cur][(wn * 64 + ni * 16 + lr) * BK];
            bf4[ni].h[0] = *(const ush8*)(p + half * 8);
            bf4[ni].h[1] = *(const ush8*)(p + 16 + half * 8);
        }
#pragma unroll
        for (int mi = 0; mi < 2; ++mi)
#pragma unroll
            for (int ni = 0; ni < 4; ++ni)
                acc[mi][ni] = __builtin_amdgcn_wmma_f32_16x16x32_bf16(
                    false, af[mi].f, false, bf4[ni].f, (short)0, acc[mi][ni],
                    false, false);
        __syncthreads();          // protect stage buffer before next overwrite
    }

    // epilogue: C VGPR r holds M=r (lanes 0-15), M=r+8 (lanes 16-31)
#pragma unroll
    for (int mi = 0; mi < 2; ++mi) {
#pragma unroll
        for (int ni = 0; ni < 4; ++ni) {
            int col = bn * BN + wn * 64 + ni * 16 + lr;
#pragma unroll
            for (int r = 0; r < 8; ++r) {
                int row = bm * BM + wm * 32 + mi * 16 + r + half * 8;
                size_t idx = (size_t)row * N + col;
                float v = acc[mi][ni][r];
                if (Cadd) v += Cadd[idx];
                if (bias) v += bias[col];
                C[idx] = act_apply(v, act);
            }
        }
    }
}

// ---------------------------------------------------------------------------
// Simple fp32 GEMM: C = act( alpha*(A@B[t]) + beta*Cadd + bias )
// ---------------------------------------------------------------------------
__global__ void sgemm_kernel(const float* __restrict__ A, const float* __restrict__ B,
                             const float* __restrict__ bias, const float* __restrict__ Cadd,
                             float* __restrict__ C, int M, int N, int K, int transB,
                             float alpha, float beta, int act)
{
    int idx = blockIdx.x * blockDim.x + threadIdx.x;
    if (idx >= M * N) return;
    int m = idx / N, n = idx - m * N;
    float s = 0.f;
    if (transB) {
        const float* a = A + (size_t)m * K;
        const float* b = B + (size_t)n * K;
        for (int k = 0; k < K; ++k) s += a[k] * b[k];
    } else {
        const float* a = A + (size_t)m * K;
        for (int k = 0; k < K; ++k) s += a[k] * B[(size_t)k * N + n];
    }
    float v = alpha * s;
    if (Cadd) v += beta * Cadd[idx];
    if (bias) v += bias[n];
    C[idx] = act_apply(v, act);
}

// ---------------------------------------------------------------------------
// Embedding + sinusoidal PE: h[s,t,d] = sum_f x[s,t,f]*W[f,d] + b[d] + pe(t,d)
// ---------------------------------------------------------------------------
__global__ void embed_kernel(const float* __restrict__ x, const float* __restrict__ W,
                             const float* __restrict__ b, float* __restrict__ h,
                             int M, int T, int F, int D)
{
    int idx = blockIdx.x * blockDim.x + threadIdx.x;
    if (idx >= M * D) return;
    int d = idx & (D - 1);
    int st = idx / D;
    int t = st & (T - 1);
    const float* xr = x + (size_t)st * F;
    float s = b[d];
#pragma unroll
    for (int f = 0; f < 16; ++f) s += xr[f] * W[f * D + d];
    int i2 = (d >> 1) << 1;
    float ang = (float)t * powf(10000.f, -(float)i2 / (float)D);
    s += (d & 1) ? cosf(ang) : sinf(ang);
    h[idx] = s;
}

// ---------------------------------------------------------------------------
// fp32 -> bf16 (round to nearest even)
// ---------------------------------------------------------------------------
__global__ void f32_to_bf16_kernel(const float* __restrict__ in,
                                   unsigned short* __restrict__ out, int n)
{
    int i = blockIdx.x * blockDim.x + threadIdx.x;
    if (i >= n) return;
    unsigned int bbits = __float_as_uint(in[i]);
    out[i] = (unsigned short)((bbits + 0x7fffu + ((bbits >> 16) & 1u)) >> 16);
}

// Transposed weight convert: out[n*Kd+k] = bf16( W[(rowOff+k)*N + n] )
__global__ void convT_bf16_kernel(const float* __restrict__ W,
                                  unsigned short* __restrict__ out,
                                  int Kd, int N, int rowOff)
{
    int idx = blockIdx.x * blockDim.x + threadIdx.x;
    if (idx >= Kd * N) return;
    int n = idx / Kd, k = idx - n * Kd;
    unsigned int bbits = __float_as_uint(W[(size_t)(rowOff + k) * N + n]);
    out[(size_t)n * Kd + k] =
        (unsigned short)((bbits + 0x7fffu + ((bbits >> 16) & 1u)) >> 16);
}

// ---------------------------------------------------------------------------
// LayerNorm (D==256): out = LN(X [+ R]) * g + b ; one wave32 per row
// ---------------------------------------------------------------------------
__global__ void ln_kernel(const float* __restrict__ X, const float* __restrict__ R,
                          const float* __restrict__ g, const float* __restrict__ b,
                          float* __restrict__ out, int rows)
{
    const int D = 256;
    int wave = threadIdx.x >> 5;
    int lane = threadIdx.x & 31;
    int row = blockIdx.x * (blockDim.x >> 5) + wave;
    if (row >= rows) return;
    const float* xr = X + (size_t)row * D;
    const float* rr = R ? R + (size_t)row * D : nullptr;
    float v[8];
    float s = 0.f;
#pragma unroll
    for (int i = 0; i < 8; ++i) {
        int d = lane + 32 * i;
        float t = xr[d] + (rr ? rr[d] : 0.f);
        v[i] = t; s += t;
    }
#pragma unroll
    for (int m = 16; m >= 1; m >>= 1) s += __shfl_xor(s, m, 32);
    float mean = s * (1.f / D);
    float q = 0.f;
#pragma unroll
    for (int i = 0; i < 8; ++i) { float t = v[i] - mean; q += t * t; }
#pragma unroll
    for (int m = 16; m >= 1; m >>= 1) q += __shfl_xor(q, m, 32);
    float inv = rsqrtf(q * (1.f / D) + 1e-5f);
    float* orow = out + (size_t)row * D;
#pragma unroll
    for (int i = 0; i < 8; ++i) {
        int d = lane + 32 * i;
        orow[d] = (v[i] - mean) * inv * g[d] + b[d];
    }
}

// ---------------------------------------------------------------------------
// Per-(stock,head) temporal self-attention, T=64, dh=32. K/V staged in LDS.
// ---------------------------------------------------------------------------
#define ATT_T 64
#define ATT_DH 32
__global__ void attn_kernel(const float* __restrict__ q, const float* __restrict__ k,
                            const float* __restrict__ v, float* __restrict__ o,
                            int D, int H)
{
    int sh = blockIdx.x;
    int s = sh / H, head = sh - (sh / H) * H;
    int t = threadIdx.x;                  // 0..63 query index
    __shared__ float ksh[ATT_T * ATT_DH];
    __shared__ float vsh[ATT_T * ATT_DH];
    size_t base = ((size_t)s * ATT_T) * D + head * ATT_DH;
#pragma unroll
    for (int d = 0; d < ATT_DH; ++d) {
        ksh[t * ATT_DH + d] = k[base + (size_t)t * D + d];
        vsh[t * ATT_DH + d] = v[base + (size_t)t * D + d];
    }
    __syncthreads();
    const float* qr = q + base + (size_t)t * D;
    float qv[ATT_DH];
#pragma unroll
    for (int d = 0; d < ATT_DH; ++d) qv[d] = qr[d];
    float sc[ATT_T];
    float mx = -1e30f;
    const float scale = 0.17677669529663687f;   // 1/sqrt(32)
    for (int j = 0; j < ATT_T; ++j) {
        float a = 0.f;
#pragma unroll
        for (int d = 0; d < ATT_DH; ++d) a += qv[d] * ksh[j * ATT_DH + d];
        a *= scale;
        sc[j] = a;
        mx = fmaxf(mx, a);
    }
    float sum = 0.f;
    for (int j = 0; j < ATT_T; ++j) { sc[j] = expf(sc[j] - mx); sum += sc[j]; }
    float inv = 1.f / sum;
    float ov[ATT_DH];
#pragma unroll
    for (int d = 0; d < ATT_DH; ++d) ov[d] = 0.f;
    for (int j = 0; j < ATT_T; ++j) {
        float a = sc[j] * inv;
#pragma unroll
        for (int d = 0; d < ATT_DH; ++d) ov[d] += a * vsh[j * ATT_DH + d];
    }
    float* orow = o + base + (size_t)t * D;
#pragma unroll
    for (int d = 0; d < ATT_DH; ++d) orow[d] = ov[d];
}

// ---------------------------------------------------------------------------
// Temporal attention pooling: a = softmax_t(u . vt); xs = sum_t a*h
// ---------------------------------------------------------------------------
__global__ void pool_kernel(const float* __restrict__ u, const float* __restrict__ h,
                            const float* __restrict__ vt, float* __restrict__ xs,
                            int T, int D)
{
    int s = blockIdx.x;
    int t = threadIdx.x;
    __shared__ float sa[ATT_T];
    __shared__ float red[2];
    const float* ur = u + ((size_t)s * T + t) * D;
    float acc = 0.f;
    for (int d = 0; d < D; ++d) acc += ur[d] * vt[d];
    sa[t] = acc;
    __syncthreads();
    if (t == 0) { float m = -1e30f; for (int j = 0; j < ATT_T; ++j) m = fmaxf(m, sa[j]); red[0] = m; }
    __syncthreads();
    float e = expf(sa[t] - red[0]);
    sa[t] = e;
    __syncthreads();
    if (t == 0) { float sm = 0.f; for (int j = 0; j < ATT_T; ++j) sm += sa[j]; red[1] = sm; }
    __syncthreads();
    sa[t] = sa[t] / red[1];
    __syncthreads();
    for (int d = t; d < D; d += blockDim.x) {
        float a2 = 0.f;
        for (int j = 0; j < ATT_T; ++j) a2 += sa[j] * h[((size_t)s * T + j) * D + d];
        xs[(size_t)s * D + d] = a2;
    }
}

// ---------------------------------------------------------------------------
// Hypergraph normalization builds
// ---------------------------------------------------------------------------
__global__ void dv_kernel(const float* __restrict__ Hinc, float* __restrict__ Dv, int S, int E)
{
    int s = blockIdx.x * blockDim.x + threadIdx.x;
    if (s >= S) return;
    float a = 1e-6f;
    for (int e = 0; e < E; ++e) a += Hinc[(size_t)s * E + e];
    Dv[s] = a;
}
__global__ void de_kernel(const float* __restrict__ Hinc, float* __restrict__ De, int S, int E)
{
    int e = blockIdx.x * blockDim.x + threadIdx.x;
    if (e >= E) return;
    float a = 1e-6f;
    for (int s = 0; s < S; ++s) a += Hinc[(size_t)s * E + e];
    De[e] = a;
}
// Hn[s,e] = Hinc * Dv^-1/2 ; HnT[e,s] = Hn[s,e]/De[e]   (A = Hn_div_De @ Hn^T)
__global__ void hn_kernel(const float* __restrict__ Hinc, const float* __restrict__ Dv,
                          const float* __restrict__ De, float* __restrict__ Hn,
                          float* __restrict__ HnT, int S, int E)
{
    int idx = blockIdx.x * blockDim.x + threadIdx.x;
    if (idx >= S * E) return;
    int s = idx / E, e = idx - s * E;
    float hn = Hinc[idx] * rsqrtf(Dv[s]);
    Hn[idx] = hn;
    HnT[(size_t)e * S + s] = hn / De[e];
}

// ---------------------------------------------------------------------------
// elementwise helpers
// ---------------------------------------------------------------------------
__global__ void axpy3_kernel(const float* __restrict__ x, const float* __restrict__ y,
                             const float* __restrict__ z, float c0, float c1, float c2,
                             float* __restrict__ out, int n)
{
    int i = blockIdx.x * blockDim.x + threadIdx.x;
    if (i >= n) return;
    out[i] = c0 * x[i] + c1 * y[i] + c2 * z[i];
}
__global__ void rowscale_kernel(const float* __restrict__ in, const float* __restrict__ rv,
                                float* __restrict__ out, int n, int D)
{
    int i = blockIdx.x * blockDim.x + threadIdx.x;
    if (i >= n) return;
    out[i] = rv[i / D] * in[i];
}
__global__ void scale_acc_kernel(const float* __restrict__ src, const float* __restrict__ scaleArr,
                                 int sidx, float* __restrict__ dst, int n)
{
    int i = blockIdx.x * blockDim.x + threadIdx.x;
    if (i >= n) return;
    dst[i] += scaleArr[sidx] * src[i];
}
__global__ void fill0_kernel(float* __restrict__ p, int n)
{
    int i = blockIdx.x * blockDim.x + threadIdx.x;
    if (i < n) p[i] = 0.f;
}
__global__ void fusion_kernel(const float* __restrict__ beta, const float* __restrict__ xs,
                              const float* __restrict__ dyn, const float* __restrict__ hyper,
                              float* __restrict__ out, int n)
{
    int i = blockIdx.x * blockDim.x + threadIdx.x;
    if (i >= n) return;
    float b0 = beta[0], b1 = beta[1], b2 = beta[2];
    float m = fmaxf(b0, fmaxf(b1, b2));
    float e0 = expf(b0 - m), e1 = expf(b1 - m), e2 = expf(b2 - m);
    float inv = 1.f / (e0 + e1 + e2);
    out[i] = (e0 * xs[i] + e1 * dyn[i] + e2 * hyper[i]) * inv;
}

// ---------------------------------------------------------------------------
// top-k threshold (k=32): kth[row] = 32nd largest of sim[row,:]
// ---------------------------------------------------------------------------
__global__ void kth_kernel(const float* __restrict__ sim, float* __restrict__ kth, int S)
{
    int row = blockIdx.x * blockDim.x + threadIdx.x;
    if (row >= S) return;
    float best[32];
#pragma unroll
    for (int i = 0; i < 32; ++i) best[i] = -1e30f;
    int minI = 0;
    const float* r = sim + (size_t)row * S;
    for (int j = 0; j < S; ++j) {
        float v = r[j];
        if (v > best[minI]) {
            best[minI] = v;
            minI = 0;
            for (int i = 1; i < 32; ++i) if (best[i] < best[minI]) minI = i;
        }
    }
    kth[row] = best[minI];
}

// masked row softmax in place: entries < kth -> 0
__global__ void masked_softmax_kernel(float* __restrict__ sim, const float* __restrict__ kth, int S)
{
    int row = blockIdx.x;
    int tid = threadIdx.x;                 // 256 threads
    __shared__ float red[256];
    float kv = kth[row];
    float* r = sim + (size_t)row * S;
    float m = -1e30f;
    for (int j = tid; j < S; j += 256) { float v = r[j]; if (v >= kv) m = fmaxf(m, v); }
    red[tid] = m; __syncthreads();
    for (int st = 128; st > 0; st >>= 1) { if (tid < st) red[tid] = fmaxf(red[tid], red[tid + st]); __syncthreads(); }
    m = red[0]; __syncthreads();
    float sum = 0.f;
    for (int j = tid; j < S; j += 256) {
        float v = r[j];
        float e = (v >= kv) ? expf(v - m) : 0.f;
        r[j] = e; sum += e;
    }
    red[tid] = sum; __syncthreads();
    for (int st = 128; st > 0; st >>= 1) { if (tid < st) red[tid] += red[tid + st]; __syncthreads(); }
    float inv = 1.f / red[0];
    for (int j = tid; j < S; j += 256) r[j] *= inv;
}

// ---------------------------------------------------------------------------
// Host orchestration
// ---------------------------------------------------------------------------
extern "C" void kernel_launch(void* const* d_in, const int* in_sizes, int n_in,
                              void* d_out, int out_size, void* d_ws, size_t ws_size,
                              hipStream_t stream)
{
    const int S = 2048, T = 64, F = 16, D = 256, NS = 4, E = 128, H = 8;
    const int M = S * T;                       // 131072 token rows
    const int MD = M * D;                      // 33.5M
    const int SD = S * D;

    const float* x      = (const float*)d_in[0];
    const float* snap   = (const float*)d_in[1];
    const float* psnap  = (const float*)d_in[2];
    const float* beta   = (const float*)d_in[3];
    const float* W_emb  = (const float*)d_in[4];
    const float* b_emb  = (const float*)d_in[5];
    const float* Wq = (const float*)d_in[6];  const float* bq = (const float*)d_in[7];
    const float* Wk = (const float*)d_in[8];  const float* bk = (const float*)d_in[9];
    const float* Wv = (const float*)d_in[10]; const float* bv = (const float*)d_in[11];
    const float* Wo = (const float*)d_in[12]; const float* bo = (const float*)d_in[13];
    const float* ln1_g = (const float*)d_in[14]; const float* ln1_b = (const float*)d_in[15];
    const float* W_int = (const float*)d_in[16]; const float* b_int = (const float*)d_in[17];
    const float* W_mix = (const float*)d_in[18]; const float* b_mix = (const float*)d_in[19];
    const float* ln2_g = (const float*)d_in[20]; const float* ln2_b = (const float*)d_in[21];
    const float* Wt = (const float*)d_in[22]; const float* bt = (const float*)d_in[23];
    const float* vt = (const float*)d_in[24];
    const float* Wh1 = (const float*)d_in[25]; const float* bh1 = (const float*)d_in[26];
    const float* filt1 = (const float*)d_in[27];
    const float* Wh2 = (const float*)d_in[28]; const float* bh2 = (const float*)d_in[29];
    const float* filt2 = (const float*)d_in[30];
    const float* Wa = (const float*)d_in[31]; const float* Wb = (const float*)d_in[32];
    const float* Wg = (const float*)d_in[33]; const float* bg = (const float*)d_in[34];
    const float* Wm1 = (const float*)d_in[35]; const float* bm1 = (const float*)d_in[36];
    const float* Wm2 = (const float*)d_in[37]; const float* bm2 = (const float*)d_in[38];
    float* out = (float*)d_out;

    // bump allocator over workspace
    char* ws = (char*)d_ws;
    size_t off = 0;
    auto alloc = [&](size_t bytes) -> void* {
        void* p = ws + off;
        off = (off + bytes + 255) & ~(size_t)255;
        return p;
    };
    const size_t MD4 = (size_t)MD * 4, MD2 = (size_t)MD * 2;
    float* hF  = (float*)alloc(MD4);           // current hidden state (S,T,D)
    float* t1F = (float*)alloc(MD4);
    float* t2F = (float*)alloc(MD4);
    float* qF  = (float*)alloc(MD4);
    float* kF  = (float*)alloc(MD4);
    float* vF  = (float*)alloc(MD4);
    unsigned short* hB = (unsigned short*)alloc(MD2);
    unsigned short* iB = (unsigned short*)alloc(MD2);
    const size_t WB = (size_t)D * D * 2;
    unsigned short* wqT   = (unsigned short*)alloc(WB);
    unsigned short* wkT   = (unsigned short*)alloc(WB);
    unsigned short* wvT   = (unsigned short*)alloc(WB);
    unsigned short* woT   = (unsigned short*)alloc(WB);
    unsigned short* wintT = (unsigned short*)alloc(WB);
    unsigned short* wmAT  = (unsigned short*)alloc(WB);
    unsigned short* wmBT  = (unsigned short*)alloc(WB);
    unsigned short* wtT   = (unsigned short*)alloc(WB);
    float* xsB   = (float*)alloc((size_t)SD * 4);
    float* hyperB= (float*)alloc((size_t)SD * 4);
    float* dynB  = (float*)alloc((size_t)SD * 4);
    float* gA    = (float*)alloc((size_t)SD * 4);
    float* gB    = (float*)alloc((size_t)SD * 4);
    float* gC    = (float*)alloc((size_t)SD * 4);
    float* hgL1  = (float*)alloc((size_t)SD * 4);
    float* hgL2  = (float*)alloc((size_t)SD * 4);
    float* hgP   = (float*)alloc((size_t)SD * 4);
    float* hgY   = (float*)alloc((size_t)SD * 4);
    float* hgR1  = (float*)alloc((size_t)SD * 4);
    float* hgOut = (float*)alloc((size_t)SD * 4);
    float* tE    = (float*)alloc((size_t)E * D * 4);
    float* HnBuf = (float*)alloc((size_t)S * E * 4);
    float* HnTBuf= (float*)alloc((size_t)E * S * 4);
    float* DvBuf = (float*)alloc((size_t)S * 4);
    float* DeBuf = (float*)alloc((size_t)E * 4);
    float* kthBuf= (float*)alloc((size_t)S * 4);
    float* simBuf= (float*)alloc((size_t)S * S * 4);

    auto g1 = [](int n) { return dim3((n + 255) / 256); };

    auto wgemm = [&](const unsigned short* A, const unsigned short* Bt, const float* bias,
                     const float* Cadd, float* C, int m, int n, int kk, int act) {
        int blocks = (m / BM) * (n / BN);
        wmma_gemm_tiled_kernel<<<dim3(blocks), dim3(256), 0, stream>>>(
            A, Bt, bias, Cadd, C, m, n, kk, act);
    };
    auto sgemm = [&](const float* A, const float* B, const float* bias, const float* Cadd,
                     float* C, int m, int n, int kk, int tB, float alpha, float bb, int act) {
        sgemm_kernel<<<g1(m * n), dim3(256), 0, stream>>>(A, B, bias, Cadd, C,
                                                          m, n, kk, tB, alpha, bb, act);
    };
    auto toBf = [&](const float* in, unsigned short* o, int n) {
        f32_to_bf16_kernel<<<g1(n), dim3(256), 0, stream>>>(in, o, n);
    };

    // ---- weight conversions (bf16, transposed to NxK) -----------------------
    convT_bf16_kernel<<<g1(D * D), dim3(256), 0, stream>>>(Wq, wqT, D, D, 0);
    convT_bf16_kernel<<<g1(D * D), dim3(256), 0, stream>>>(Wk, wkT, D, D, 0);
    convT_bf16_kernel<<<g1(D * D), dim3(256), 0, stream>>>(Wv, wvT, D, D, 0);
    convT_bf16_kernel<<<g1(D * D), dim3(256), 0, stream>>>(Wo, woT, D, D, 0);
    convT_bf16_kernel<<<g1(D * D), dim3(256), 0, stream>>>(W_int, wintT, D, D, 0);
    convT_bf16_kernel<<<g1(D * D), dim3(256), 0, stream>>>(W_mix, wmAT, D, D, 0);
    convT_bf16_kernel<<<g1(D * D), dim3(256), 0, stream>>>(W_mix, wmBT, D, D, D);
    convT_bf16_kernel<<<g1(D * D), dim3(256), 0, stream>>>(Wt, wtT, D, D, 0);

    // ---- 1) embedding + PE --------------------------------------------------
    embed_kernel<<<g1(MD), dim3(256), 0, stream>>>(x, W_emb, b_emb, hF, M, T, F, D);
    toBf(hF, hB, MD);

    // ---- 2) Q,K,V projections (WMMA) ---------------------------------------
    wgemm(hB, wqT, bq, nullptr, qF, M, D, D, 0);
    wgemm(hB, wkT, bk, nullptr, kF, M, D, D, 0);
    wgemm(hB, wvT, bv, nullptr, vF, M, D, D, 0);

    // ---- 3) multi-head temporal attention ----------------------------------
    attn_kernel<<<dim3(S * H), dim3(ATT_T), 0, stream>>>(qF, kF, vF, t1F, D, H);

    // ---- 4) output projection + residual + LN1 -----------------------------
    toBf(t1F, iB, MD);
    wgemm(iB, woT, bo, nullptr, t2F, M, D, D, 0);
    ln_kernel<<<dim3(M / 8), dim3(256), 0, stream>>>(hF, t2F, ln1_g, ln1_b, hF, M);

    // ---- 5) intensity + mix + LN2 ------------------------------------------
    toBf(hF, hB, MD);
    wgemm(hB, wintT, b_int, nullptr, t1F, M, D, D, 4);              // softplus
    toBf(t1F, iB, MD);
    wgemm(hB, wmAT, nullptr, nullptr, t2F, M, D, D, 0);
    wgemm(iB, wmBT, b_mix, t2F, t1F, M, D, D, 1);                   // relu
    ln_kernel<<<dim3(M / 8), dim3(256), 0, stream>>>(t1F, nullptr, ln2_g, ln2_b, hF, M);

    // ---- 6) temporal attention pooling -------------------------------------
    toBf(hF, hB, MD);
    wgemm(hB, wtT, bt, nullptr, t1F, M, D, D, 3);                   // tanh
    pool_kernel<<<dim3(S), dim3(ATT_T), 0, stream>>>(t1F, hF, vt, xsB, T, D);

    // ---- 7) wavelet hypergraph conv per snapshot ---------------------------
    // L v = v - Hn_div_De (Hn^T v); psi(v)=v-Lv+0.5 L^2 v; psi_inv(v)=v+Lv+0.5 L^2 v
    fill0_kernel<<<g1(SD), dim3(256), 0, stream>>>(hyperB, SD);
    auto applyL = [&](const float* vin, float* vout) {
        sgemm(HnTBuf, vin, nullptr, nullptr, tE, E, D, S, 0, 1.f, 0.f, 0);     // (E,D)
        sgemm(HnBuf, tE, nullptr, vin, vout, S, D, E, 0, -1.f, 1.f, 0);        // v - A v
    };
    auto whconv = [&](const float* zin, const float* filt, const float* W,
                      const float* bias, int act, float* zout) {
        applyL(zin, hgL1);
        applyL(hgL1, hgL2);
        axpy3_kernel<<<g1(SD), dim3(256), 0, stream>>>(zin, hgL1, hgL2, 1.f, 1.f, 0.5f, hgP, SD);
        rowscale_kernel<<<g1(SD), dim3(256), 0, stream>>>(hgP, filt, hgY, SD, D);
        applyL(hgY, hgL1);
        applyL(hgL1, hgL2);
        axpy3_kernel<<<g1(SD), dim3(256), 0, stream>>>(hgY, hgL1, hgL2, 1.f, -1.f, 0.5f, hgP, SD);
        sgemm(hgP, W, bias, nullptr, zout, S, D, D, 0, 1.f, 0.f, act);
    };
    for (int n = 0; n < NS; ++n) {
        const float* Hinc = snap + (size_t)n * S * E;
        dv_kernel<<<g1(S), dim3(256), 0, stream>>>(Hinc, DvBuf, S, E);
        de_kernel<<<g1(E), dim3(256), 0, stream>>>(Hinc, DeBuf, S, E);
        hn_kernel<<<g1(S * E), dim3(256), 0, stream>>>(Hinc, DvBuf, DeBuf, HnBuf, HnTBuf, S, E);
        whconv(xsB, filt1, Wh1, bh1, 2, hgR1);                      // leaky relu 0.1
        whconv(hgR1, filt2, Wh2, bh2, 0, hgOut);
        scale_acc_kernel<<<g1(SD), dim3(256), 0, stream>>>(hgOut, psnap, n, hyperB, SD);
    }

    // ---- 8) dynamic top-k graph --------------------------------------------
    sgemm(xsB, Wa, nullptr, nullptr, gA, S, D, D, 0, 1.f, 0.f, 0);  // P = xs@Wa
    sgemm(xsB, Wb, nullptr, nullptr, gB, S, D, D, 0, 1.f, 0.f, 0);  // Q = xs@Wb
    sgemm(gA, gB, nullptr, nullptr, simBuf, S, S, D, 1, 1.f / 16.f, 0.f, 0);  // /sqrt(256)
    kth_kernel<<<g1(S), dim3(256), 0, stream>>>(simBuf, kthBuf, S);
    masked_softmax_kernel<<<dim3(S), dim3(256), 0, stream>>>(simBuf, kthBuf, S);
    sgemm(simBuf, xsB, nullptr, nullptr, gC, S, D, S, 0, 1.f, 0.f, 0);        // Adyn@xs
    sgemm(gC, Wg, bg, nullptr, dynB, S, D, D, 0, 1.f, 0.f, 1);                // relu

    // ---- 9) fusion + MLP head ----------------------------------------------
    fusion_kernel<<<g1(SD), dim3(256), 0, stream>>>(beta, xsB, dynB, hyperB, gA, SD);
    sgemm(gA, Wm1, bm1, nullptr, gC, S, D, D, 0, 1.f, 0.f, 1);                // relu
    sgemm(gC, Wm2, bm2, nullptr, out, S, 1, D, 0, 1.f, 0.f, 0);               // (S,1)
}
